// GatedGCNConv_25598005084171
// MI455X (gfx1250) — compile-verified, matrix operand order checked by reading, not run on hardware
//
#include <hip/hip_runtime.h>

// ---------------------------------------------------------------------------
// Types for CDNA5 WMMA
// ---------------------------------------------------------------------------
typedef __attribute__((ext_vector_type(16))) __bf16 v16bf;
typedef __attribute__((ext_vector_type(8)))  float  v8f;

#define WMMA_BF16(A, B, C) \
    __builtin_amdgcn_wmma_f32_16x16x32_bf16(false, (A), false, (B), (short)0, (C), false, false)

#define NSLOT 64  // stat-accumulation slots (reduce atomic contention)

// Load 16 floats (p[0..7] and p[16..23]) and split into bf16 hi/lo fragments
// using native hardware converts (v_cvt_pk_bf16_f32 on gfx1250).
// Matches the 16-bit A/B fragment layout: elements 0..7 -> K base..base+7,
// elements 8..15 -> K base+16..base+23.
__device__ __forceinline__ void load_frag_f32(const float* __restrict__ p,
                                              v16bf& hi, v16bf& lo) {
#pragma unroll
    for (int i = 0; i < 8; ++i) {
        float x = p[i];
        __bf16 h = (__bf16)x;
        hi[i] = h;
        lo[i] = (__bf16)(x - (float)h);
        float y = p[16 + i];
        __bf16 g = (__bf16)y;
        hi[8 + i] = g;
        lo[8 + i] = (__bf16)(y - (float)g);
    }
}

// Fetch a pre-packed weight fragment: 32 contiguous bf16 per (fragment, lane):
// [0..15] = hi, [16..31] = lo.
__device__ __forceinline__ void load_frag_packed(const __bf16* __restrict__ pw,
                                                 int kb, int wt, int lane,
                                                 v16bf& hi, v16bf& lo) {
    const __bf16* q = pw + (((size_t)(kb * 8 + wt) * 32 + lane) << 5);
    hi = *(const v16bf*)q;
    lo = *(const v16bf*)(q + 16);
}

// ---------------------------------------------------------------------------
// Kernel 0: pack one 128x128 weight matrix into WMMA fragment order (hi/lo).
// 1024 work items: t = (kb*8 + ct)*32 + lane.
// ---------------------------------------------------------------------------
__global__ __launch_bounds__(256)
void pack_w_kernel(const float* __restrict__ W, __bf16* __restrict__ out) {
    const int t = blockIdx.x * 256 + threadIdx.x;
    if (t >= 1024) return;
    const int lane = t & 31;
    const int fid  = t >> 5;          // kb*8 + ct
    const int ct   = fid & 7;
    const int kb   = fid >> 3;
    const int m    = lane & 15;
    const int half = lane >> 4;
    const int col  = ct * 16 + m;     // output feature = W row
    const int koff = kb * 32 + half * 8;

    const float* p = W + (size_t)col * 128 + koff;
    __bf16* o = out + ((size_t)t << 5);
#pragma unroll
    for (int i = 0; i < 8; ++i) {
        float x = p[i];
        __bf16 h = (__bf16)x;
        o[i]      = h;
        o[16 + i] = (__bf16)(x - (float)h);
        float y = p[16 + i];
        __bf16 g = (__bf16)y;
        o[8 + i]  = g;
        o[24 + i] = (__bf16)(y - (float)g);
    }
}

// ---------------------------------------------------------------------------
// Kernel 1: node GEMMs  Ah/Bh/Dh/Eh = feat @ W{A,B,D,E}.T + b
// 256 threads = 8 waves; wave w owns column tile w (16 cols); block owns 16 rows.
// bf16x3 split product per K block: Ahi*Bhi + Ahi*Blo + Alo*Bhi (fp32 acc).
// ---------------------------------------------------------------------------
__global__ __launch_bounds__(256)
void node_gemm4_kernel(const float* __restrict__ feat,
                       const __bf16* __restrict__ pWA, const float* __restrict__ bA,
                       const __bf16* __restrict__ pWB, const float* __restrict__ bB,
                       const __bf16* __restrict__ pWD, const float* __restrict__ bD,
                       const __bf16* __restrict__ pWE, const float* __restrict__ bE,
                       float* __restrict__ Ah, float* __restrict__ Bh,
                       float* __restrict__ Dh, float* __restrict__ Eh, int N) {
    const int tid  = threadIdx.x;
    const int lane = tid & 31;
    const int wt   = tid >> 5;       // 0..7 column tile
    const int m    = lane & 15;
    const int half = lane >> 4;
    const int row0 = blockIdx.x * 16;

    const int arow = (row0 + m < N) ? (row0 + m) : (N - 1);
    const int col  = wt * 16 + m;

    const float* ap = feat + (size_t)arow * 128;

    v8f acc0 = {}, acc1 = {}, acc2 = {}, acc3 = {};

#pragma unroll
    for (int kb = 0; kb < 4; ++kb) {
        const int koff = kb * 32 + half * 8;
        v16bf ahi, alo, bhi, blo;
        load_frag_f32(ap + koff, ahi, alo);

        load_frag_packed(pWA, kb, wt, lane, bhi, blo);
        acc0 = WMMA_BF16(ahi, bhi, acc0);
        acc0 = WMMA_BF16(ahi, blo, acc0);
        acc0 = WMMA_BF16(alo, bhi, acc0);

        load_frag_packed(pWB, kb, wt, lane, bhi, blo);
        acc1 = WMMA_BF16(ahi, bhi, acc1);
        acc1 = WMMA_BF16(ahi, blo, acc1);
        acc1 = WMMA_BF16(alo, bhi, acc1);

        load_frag_packed(pWD, kb, wt, lane, bhi, blo);
        acc2 = WMMA_BF16(ahi, bhi, acc2);
        acc2 = WMMA_BF16(ahi, blo, acc2);
        acc2 = WMMA_BF16(alo, bhi, acc2);

        load_frag_packed(pWE, kb, wt, lane, bhi, blo);
        acc3 = WMMA_BF16(ahi, bhi, acc3);
        acc3 = WMMA_BF16(ahi, blo, acc3);
        acc3 = WMMA_BF16(alo, bhi, acc3);
    }

    const float ba = bA[col], bb = bB[col], bd = bD[col], be = bE[col];
#pragma unroll
    for (int r = 0; r < 8; ++r) {
        const int row = row0 + r + half * 8;
        if (row < N) {
            const size_t idx = (size_t)row * 128 + col;
            Ah[idx] = acc0[r] + ba;
            Bh[idx] = acc1[r] + bb;
            Dh[idx] = acc2[r] + bd;
            Eh[idx] = acc3[r] + be;
        }
    }
}

// ---------------------------------------------------------------------------
// Kernel 2: fused edge pipeline
//   Ce = edge_feat @ WC.T + bC                           (WMMA)
//   e_pre = Dh[src] + Eh[dst] + Ce ; sigma = sigmoid(e_pre)
//   e_pre -> d_out (e region); atomic scatter of sigma and Bh[src]*sigma;
//   per-column BN-e stat partials into slots.
// ---------------------------------------------------------------------------
__global__ __launch_bounds__(256)
void edge_fused_kernel(const float* __restrict__ edge_feat,
                       const __bf16* __restrict__ pWC, const float* __restrict__ bC,
                       const float* __restrict__ Dh, const float* __restrict__ Eh,
                       const float* __restrict__ Bh,
                       const int* __restrict__ src, const int* __restrict__ dst,
                       float* __restrict__ e_pre,
                       float* __restrict__ sum_sh, float* __restrict__ sum_s,
                       float* __restrict__ slots_e, int E) {
    const int tid  = threadIdx.x;
    const int lane = tid & 31;
    const int wt   = tid >> 5;
    const int m    = lane & 15;
    const int half = lane >> 4;
    const int row0 = blockIdx.x * 16;

    const int arow = (row0 + m < E) ? (row0 + m) : (E - 1);
    const int col  = wt * 16 + m;

    const float* ap = edge_feat + (size_t)arow * 128;

    v8f acc = {};
#pragma unroll
    for (int kb = 0; kb < 4; ++kb) {
        const int koff = kb * 32 + half * 8;
        v16bf ahi, alo, bhi, blo;
        load_frag_f32(ap + koff, ahi, alo);
        load_frag_packed(pWC, kb, wt, lane, bhi, blo);
        acc = WMMA_BF16(ahi, bhi, acc);
        acc = WMMA_BF16(ahi, blo, acc);
        acc = WMMA_BF16(alo, bhi, acc);
    }

    const float bias = bC[col];
    float s = 0.f, sq = 0.f;
#pragma unroll
    for (int r = 0; r < 8; ++r) {
        const int erow = row0 + r + half * 8;
        if (erow < E) {
            const int sn = src[erow];
            const int dn = dst[erow];
            const size_t si = (size_t)sn * 128 + col;
            const size_t di = (size_t)dn * 128 + col;
            const float ep = acc[r] + bias + Dh[si] + Eh[di];
            const float sg = 1.0f / (1.0f + __expf(-ep));
            e_pre[(size_t)erow * 128 + col] = ep;
            atomicAdd(&sum_sh[di], Bh[si] * sg);
            atomicAdd(&sum_s[di], sg);
            s += ep;
            sq += ep * ep;
        }
    }

    // Combine the two half-lanes sharing a column, then one atomic per column.
    __shared__ float ls[256], lq[256];
    ls[tid] = s; lq[tid] = sq;
    __syncthreads();
    if (lane < 16) {
        const float cs = ls[tid] + ls[tid + 16];
        const float cq = lq[tid] + lq[tid + 16];
        const int slot = (blockIdx.x & (NSLOT - 1)) * 256;
        atomicAdd(&slots_e[slot + col], cs);
        atomicAdd(&slots_e[slot + 128 + col], cq);
    }
}

// ---------------------------------------------------------------------------
// Kernel 3: h_pre = Ah + sum_sh / (sum_s + eps)  + BN-h stat partials
// ---------------------------------------------------------------------------
__global__ __launch_bounds__(256)
void h_combine_kernel(const float* __restrict__ Ah,
                      const float* __restrict__ sum_sh, const float* __restrict__ sum_s,
                      float* __restrict__ hout, float* __restrict__ slots_h, int N) {
    const int tid   = threadIdx.x;
    const int col   = tid & 127;
    const int rhalf = tid >> 7;
    const int row0  = blockIdx.x * 16;

    float s = 0.f, sq = 0.f;
#pragma unroll
    for (int r = rhalf; r < 16; r += 2) {
        const int row = row0 + r;
        if (row < N) {
            const size_t idx = (size_t)row * 128 + col;
            const float hp = Ah[idx] + sum_sh[idx] / (sum_s[idx] + 1e-6f);
            hout[idx] = hp;
            s += hp;
            sq += hp * hp;
        }
    }
    __shared__ float ls[256], lq[256];
    ls[tid] = s; lq[tid] = sq;
    __syncthreads();
    if (tid < 128) {
        const float cs = ls[tid] + ls[tid + 128];
        const float cq = lq[tid] + lq[tid + 128];
        const int slot = (blockIdx.x & (NSLOT - 1)) * 256;
        atomicAdd(&slots_h[slot + col], cs);
        atomicAdd(&slots_h[slot + 128 + col], cq);
    }
}

// ---------------------------------------------------------------------------
// Kernel 4: finalize BN stats -> mu/rstd for nodes (cols 0..127) & edges
// mu_rstd layout: [0..127]=mu_n [128..255]=rstd_n [256..383]=mu_e [384..511]=rstd_e
// ---------------------------------------------------------------------------
__global__ __launch_bounds__(256)
void finalize_stats_kernel(const float* __restrict__ slots_h,
                           const float* __restrict__ slots_e,
                           float* __restrict__ mu_rstd, int N, int E) {
    const int t = threadIdx.x;
    if (t < 128) {
        float s = 0.f, q = 0.f;
        for (int k = 0; k < NSLOT; ++k) {
            s += slots_h[k * 256 + t];
            q += slots_h[k * 256 + 128 + t];
        }
        const float invn = 1.0f / (float)N;
        const float mu = s * invn;
        const float var = q * invn - mu * mu;
        mu_rstd[t]       = mu;
        mu_rstd[128 + t] = rsqrtf(var + 1e-5f);
    } else {
        const int c = t - 128;
        float s = 0.f, q = 0.f;
        for (int k = 0; k < NSLOT; ++k) {
            s += slots_e[k * 256 + c];
            q += slots_e[k * 256 + 128 + c];
        }
        const float inve = 1.0f / (float)E;
        const float mu = s * inve;
        const float var = q * inve - mu * mu;
        mu_rstd[256 + c] = mu;
        mu_rstd[384 + c] = rsqrtf(var + 1e-5f);
    }
}

// ---------------------------------------------------------------------------
// Kernel 5: out = resid + relu(gamma*(out-mu)*rstd + beta)   (in place)
// ---------------------------------------------------------------------------
__global__ __launch_bounds__(256)
void norm_relu_res_kernel(const float* __restrict__ resid,
                          const float* __restrict__ gamma, const float* __restrict__ beta,
                          const float* __restrict__ mu, const float* __restrict__ rstd,
                          float* __restrict__ out, size_t total) {
    size_t i = (size_t)blockIdx.x * blockDim.x + threadIdx.x;
    const size_t stride = (size_t)gridDim.x * blockDim.x;
    for (; i < total; i += stride) {
        const int c = (int)(i & 127);
        const float v = gamma[c] * (out[i] - mu[c]) * rstd[c] + beta[c];
        out[i] = resid[i] + fmaxf(v, 0.0f);
    }
}

// ---------------------------------------------------------------------------
// Launch
// ---------------------------------------------------------------------------
extern "C" void kernel_launch(void* const* d_in, const int* in_sizes, int n_in,
                              void* d_out, int out_size, void* d_ws, size_t ws_size,
                              hipStream_t stream) {
    (void)n_in; (void)out_size; (void)ws_size;

    const float* feat      = (const float*)d_in[0];
    const float* edge_feat = (const float*)d_in[1];
    const int*   src       = (const int*)d_in[2];
    const int*   dst       = (const int*)d_in[3];
    const float* WA = (const float*)d_in[4];  const float* bA = (const float*)d_in[5];
    const float* WB = (const float*)d_in[6];  const float* bB = (const float*)d_in[7];
    const float* WC = (const float*)d_in[8];  const float* bC = (const float*)d_in[9];
    const float* WD = (const float*)d_in[10]; const float* bD = (const float*)d_in[11];
    const float* WE = (const float*)d_in[12]; const float* bE = (const float*)d_in[13];
    const float* gamma_n = (const float*)d_in[14]; const float* beta_n = (const float*)d_in[15];
    const float* gamma_e = (const float*)d_in[16]; const float* beta_e = (const float*)d_in[17];

    const int N = in_sizes[0] / 128;
    const int E = in_sizes[1] / 128;
    const size_t ND = (size_t)N * 128;
    const size_t ED = (size_t)E * 128;

    float* hout = (float*)d_out;        // [N,128] pre-BN h, then final h
    float* eout = hout + ND;            // [E,128] pre-BN e, then final e

    // Workspace layout (floats)
    float* Ah      = (float*)d_ws;
    float* Bh      = Ah + ND;
    float* Dh      = Bh + ND;
    float* Eh      = Dh + ND;
    float* sum_sh  = Eh + ND;
    float* sum_s   = sum_sh + ND;
    float* slots_h = sum_s + ND;                      // NSLOT*256
    float* slots_e = slots_h + (size_t)NSLOT * 256;   // NSLOT*256
    float* mu_rstd = slots_e + (size_t)NSLOT * 256;   // 512
    // Packed weights: 5 matrices * 1024 lanes-frags * 32 bf16 = 32768 bf16 each
    __bf16* pW  = (__bf16*)(mu_rstd + 512);
    __bf16* pWA = pW;
    __bf16* pWB = pWA + 32768;
    __bf16* pWC = pWB + 32768;
    __bf16* pWD = pWC + 32768;
    __bf16* pWE = pWD + 32768;

    // Zero scatter accumulators + stat slots (graph-capturable memset node)
    hipMemsetAsync(sum_sh, 0, (2 * ND + 2 * (size_t)NSLOT * 256) * sizeof(float), stream);

    // Pack weights into WMMA fragment order (hi/lo bf16 split)
    pack_w_kernel<<<4, 256, 0, stream>>>(WA, pWA);
    pack_w_kernel<<<4, 256, 0, stream>>>(WB, pWB);
    pack_w_kernel<<<4, 256, 0, stream>>>(WC, pWC);
    pack_w_kernel<<<4, 256, 0, stream>>>(WD, pWD);
    pack_w_kernel<<<4, 256, 0, stream>>>(WE, pWE);

    const int nblk = (N + 15) / 16;
    const int eblk = (E + 15) / 16;

    node_gemm4_kernel<<<nblk, 256, 0, stream>>>(feat, pWA, bA, pWB, bB, pWD, bD, pWE, bE,
                                                Ah, Bh, Dh, Eh, N);

    edge_fused_kernel<<<eblk, 256, 0, stream>>>(edge_feat, pWC, bC, Dh, Eh, Bh, src, dst,
                                                eout, sum_sh, sum_s, slots_e, E);

    h_combine_kernel<<<nblk, 256, 0, stream>>>(Ah, sum_sh, sum_s, hout, slots_h, N);

    finalize_stats_kernel<<<1, 256, 0, stream>>>(slots_h, slots_e, mu_rstd, N, E);

    norm_relu_res_kernel<<<4096, 256, 0, stream>>>(feat, gamma_n, beta_n,
                                                   mu_rstd, mu_rstd + 128, hout, ND);
    norm_relu_res_kernel<<<4096, 256, 0, stream>>>(edge_feat, gamma_e, beta_e,
                                                   mu_rstd + 256, mu_rstd + 384, eout, ED);
}
